// SelfAttention_26010321945311
// MI455X (gfx1250) — matrix operand order repaired and checked by reading, not compiled, not run on hardware
//
#include <hip/hip_runtime.h>
#include <hip/hip_bf16.h>

typedef __attribute__((ext_vector_type(16))) __bf16 v16bf;
typedef __attribute__((ext_vector_type(8)))  __bf16 v8bf;
typedef __attribute__((ext_vector_type(8)))  float  v8f;

#define BATCH 32
#define SEQ   1024
#define DIM   256
#define ROWS_PER_WAVE 16
#define WAVES 4
// padded LDS row stride for S (floats): keeps float2 alignment, rotates banks
#define SSTRIDE 1026
// padded row stride for transposed bf16 Q tile: 40*2B = 80B = 20 dwords.
// gcd(20,64)=4 and 16*20 mod 64 cycles through 16 distinct bank starts,
// spacing >= 4 dwords -> conflict-free ds_load_b128 across a 16-lane half.
#define QT_STRIDE 40

__global__ __launch_bounds__(128) void attn_fused_kernel(
    const float* __restrict__ q,
    float* __restrict__ out,
    float* __restrict__ attn)
{
    extern __shared__ char smem[];
    float*  Sbase = (float*)smem;  // WAVES * 16 * SSTRIDE floats (~257 KB)
    __bf16* qT    = (__bf16*)(smem + (size_t)WAVES * ROWS_PER_WAVE * SSTRIDE * sizeof(float)); // 256*40 bf16 (20 KB)

    const int b       = blockIdx.y;
    const int rowTile = blockIdx.x;                 // 0..15 (64 rows each)
    const int tid     = threadIdx.x;
    const int wave    = tid >> 5;                   // 0..3
    const int lane    = tid & 31;
    const int half    = lane >> 4;                  // 0 or 1 (lane group)
    const int l15     = lane & 15;

    const int rowBase = rowTile * 64 + wave * ROWS_PER_WAVE;
    const float* __restrict__ qb = q + (size_t)b * SEQ * DIM;
    float* __restrict__ Srow = Sbase + (size_t)wave * ROWS_PER_WAVE * SSTRIDE;

    // ---------------------------------------------------------------
    // Preload A fragments: this wave's 16 rows of Q, K=256 in 8 chunks
    // A k order used here (contiguous; B matches): lanes0-15 M=lane, K in {0..15}+chunk;
    // lanes16-31 M=lane-16, K in {16..31}+chunk.
    // ---------------------------------------------------------------
    v16bf afrag[8];
    {
        const float* qrow = qb + (size_t)(rowBase + l15) * DIM;
        #pragma unroll
        for (int kc = 0; kc < 8; ++kc) {
            const int kb = kc * 32 + half * 16;
            #pragma unroll
            for (int j = 0; j < 8; ++j) {
                const int k = kb + 2 * j;
                float2 f = *(const float2*)(qrow + k);
                afrag[kc][2 * j]     = (__bf16)f.x;
                afrag[kc][2 * j + 1] = (__bf16)f.y;
            }
        }
    }

    // ---------------------------------------------------------------
    // GEMM1: S = Q * Q^T  (16 rows x 1024 cols, fp32 accum) -> LDS
    // B k order used here (contiguous, as A): lane holds column n=l15, K=half*16 + 0..15
    // ---------------------------------------------------------------
    #pragma unroll 1
    for (int ct = 0; ct < 64; ++ct) {
        const int colBase = ct * 16;
        const float* qcol = qb + (size_t)(colBase + l15) * DIM;
        if (ct < 63) __builtin_prefetch(qcol + 16 * DIM, 0, 1);  // next col tile
        v8f acc = {};
        #pragma unroll
        for (int kc = 0; kc < 8; ++kc) {
            v16bf bfrag;
            const int k0 = kc * 32 + half * 16;
            #pragma unroll
            for (int j = 0; j < 8; ++j) {
                float2 f = *(const float2*)(qcol + k0 + 2 * j);
                bfrag[2 * j]     = (__bf16)f.x;
                bfrag[2 * j + 1] = (__bf16)f.y;
            }
            acc = __builtin_amdgcn_wmma_f32_16x16x32_bf16(
                false, afrag[kc], false, bfrag, (short)0, acc, false, false);
        }
        // C/D layout: lane column n=l15, VGPR j -> row j + half*8
        #pragma unroll
        for (int j = 0; j < 8; ++j) {
            Srow[(j + half * 8) * SSTRIDE + colBase + l15] = acc[j];
        }
    }
    __syncthreads();

    // ---------------------------------------------------------------
    // Softmax per row: mask(S==0 -> -inf), /16, max, exp, sum, NaN->0.
    // 32 lanes cooperate per row (coalesced LDS + global traffic).
    // Normalized probs written back to LDS (for GEMM2) and to attn out.
    // ---------------------------------------------------------------
    const float NEGINF = -__builtin_inff();
    #pragma unroll 1
    for (int r = 0; r < ROWS_PER_WAVE; ++r) {
        float* row = Srow + r * SSTRIDE;
        float m = NEGINF;
        #pragma unroll 4
        for (int i = lane; i < SEQ; i += 32) {
            float v  = row[i];
            float vs = (v == 0.0f) ? NEGINF : v * 0.0625f;
            m = fmaxf(m, vs);
        }
        #pragma unroll
        for (int off = 16; off > 0; off >>= 1)
            m = fmaxf(m, __shfl_xor(m, off, 32));

        float s = 0.0f;
        #pragma unroll 4
        for (int i = lane; i < SEQ; i += 32) {
            float v  = row[i];
            float vs = (v == 0.0f) ? NEGINF : v * 0.0625f;
            float e  = (m == NEGINF) ? 0.0f : __expf(vs - m);
            row[i] = e;
            s += e;
        }
        #pragma unroll
        for (int off = 16; off > 0; off >>= 1)
            s += __shfl_xor(s, off, 32);

        const float inv = (s > 0.0f) ? (1.0f / s) : 0.0f;
        float* arow = attn + ((size_t)b * SEQ + rowBase + r) * SEQ;
        #pragma unroll 4
        for (int i = lane; i < SEQ; i += 32) {
            float p = row[i] * inv;
            row[i]  = p;
            arow[i] = p;   // coalesced fp32 store of attention matrix
        }
    }
    __syncthreads();

    // ---------------------------------------------------------------
    // GEMM2: O = P * Q  (16 x 256), K = 1024 over probability columns.
    // Per k-chunk: cooperatively stage 32x256 of Q into LDS, transposed
    // to bf16 (shared by all 4 waves), then B' tiles come from two
    // ds_load_b128 per lane. All 16 output tiles stay live in VGPRs.
    // ---------------------------------------------------------------
    v8f oacc[16];
    #pragma unroll
    for (int t = 0; t < 16; ++t) oacc[t] = (v8f){};

    #pragma unroll 1
    for (int kc = 0; kc < 32; ++kc) {
        // --- cooperative transpose-stage: qT[col][k] = bf16(q[kc*32+k][col])
        {
            const float* qk = qb + (size_t)kc * 32 * DIM;
            #pragma unroll
            for (int it = 0; it < 16; ++it) {
                const int idx = it * 128 + tid;   // 0..2047
                const int n   = idx >> 6;         // 0..31 (k within chunk)
                const int c   = (idx & 63) * 4;   // col, float4-aligned
                float4 v = *(const float4*)(qk + n * DIM + c);
                qT[(c + 0) * QT_STRIDE + n] = (__bf16)v.x;
                qT[(c + 1) * QT_STRIDE + n] = (__bf16)v.y;
                qT[(c + 2) * QT_STRIDE + n] = (__bf16)v.z;
                qT[(c + 3) * QT_STRIDE + n] = (__bf16)v.w;
            }
            if (kc < 31) __builtin_prefetch(qk + 32 * DIM + tid * 16, 0, 1);
        }
        __syncthreads();

        // A' fragment from LDS P (padded stride -> conflict-free)
        v16bf ap;
        {
            const int kb = kc * 32 + half * 16;
            const float* prow = Srow + (size_t)l15 * SSTRIDE;
            #pragma unroll
            for (int j = 0; j < 8; ++j) {
                const int k = kb + 2 * j;
                float2 f = *(const float2*)(prow + k);
                ap[2 * j]     = (__bf16)f.x;
                ap[2 * j + 1] = (__bf16)f.y;
            }
        }

        #pragma unroll
        for (int t = 0; t < 16; ++t) {
            // B' k order used here (contiguous, as A'): lane holds out-column t*16+l15, K=half*16+0..15
            const __bf16* src = qT + (size_t)(t * 16 + l15) * QT_STRIDE + half * 16;
            v8bf lo = *(const v8bf*)(src);       // ds_load_b128
            v8bf hi = *(const v8bf*)(src + 8);   // ds_load_b128
            v16bf bp = __builtin_shufflevector(lo, hi,
                0, 1, 2, 3, 4, 5, 6, 7, 8, 9, 10, 11, 12, 13, 14, 15);
            oacc[t] = __builtin_amdgcn_wmma_f32_16x16x32_bf16(
                false, ap, false, bp, (short)0, oacc[t], false, false);
        }
        __syncthreads();   // qT reused next iteration
    }

    // Write output tile: lane column n, VGPR j -> row j + half*8
    #pragma unroll
    for (int t = 0; t < 16; ++t) {
        const int n = t * 16 + l15;
        #pragma unroll
        for (int j = 0; j < 8; ++j) {
            const int mrow = rowBase + j + half * 8;
            out[((size_t)b * SEQ + mrow) * DIM + n] = oacc[t][j];
        }
    }
}

extern "C" void kernel_launch(void* const* d_in, const int* in_sizes, int n_in,
                              void* d_out, int out_size, void* d_ws, size_t ws_size,
                              hipStream_t stream) {
    (void)in_sizes; (void)n_in; (void)out_size; (void)d_ws; (void)ws_size;
    const float* q = (const float*)d_in[0];
    float* out  = (float*)d_out;                                   // (32,1024,256)
    float* attn = out + (size_t)BATCH * SEQ * DIM;                 // (32,1024,1024)

    dim3 grid(16, BATCH);   // 16 row-tiles of 64 rows x 32 batches
    dim3 block(128);        // 4 waves (wave32)
    size_t shmem = (size_t)WAVES * ROWS_PER_WAVE * SSTRIDE * sizeof(float)
                 + (size_t)DIM * QT_STRIDE * sizeof(__bf16);       // ~283 KB of 320 KB/WGP
    hipLaunchKernelGGL(attn_fused_kernel, grid, block, shmem, stream, q, out, attn);
}
